// RNA_feature_extraction_72713796322178
// MI455X (gfx1250) — matrix operand order repaired and checked by reading, not compiled, not run on hardware
//
#include <hip/hip_runtime.h>

typedef __attribute__((ext_vector_type(16))) __bf16 bf16x16;
typedef __attribute__((ext_vector_type(8)))  float  f32x8;

#define LEAKY 0.2f

// ---------------------------------------------------------------------------
// helpers
// ---------------------------------------------------------------------------
static __device__ __forceinline__ unsigned ord_enc(float x) {
  unsigned u = __float_as_uint(x);
  return (u & 0x80000000u) ? ~u : (u | 0x80000000u);
}
static __device__ __forceinline__ float ord_dec(unsigned v) {
  unsigned u = (v & 0x80000000u) ? (v & 0x7FFFFFFFu) : ~v;
  return __uint_as_float(u);
}
static __device__ __forceinline__ void edge_sd(const int* ei, long E0, long e,
                                               int& s, int& d) {
  if (e < E0) { s = ei[e]; d = ei[E0 + e]; }
  else        { s = d = (int)(e - E0); }
}

// A fragment: 16x32 bf16 tile from row-major A (lda elems), per gfx1250 layout.
static __device__ __forceinline__ bf16x16 load_a_frag(const __bf16* A, int lda,
                                                      int row0, int k0, int Mlim) {
  int lane = threadIdx.x & 31;
  int m = lane & 15, hi = lane >> 4;
  int r = row0 + m; if (r >= Mlim) r = Mlim - 1;
  const __bf16* p = A + (long)r * lda + k0 + hi * 8;
  bf16x16 a;
#pragma unroll
  for (int j = 0; j < 8; ++j) a[j] = p[j];
#pragma unroll
  for (int j = 0; j < 8; ++j) a[8 + j] = p[16 + j];
  return a;
}

// B fragment: 32x16 bf16 tile, from pre-transposed Bt (N x K row-major).
static __device__ __forceinline__ bf16x16 load_b_frag(const __bf16* Bt, int ldb,
                                                      int col0, int k0) {
  int lane = threadIdx.x & 31;
  int n = lane & 15, hi = lane >> 4;
  const __bf16* p = Bt + (long)(col0 + n) * ldb + k0 + hi * 8;
  bf16x16 b;
#pragma unroll
  for (int j = 0; j < 8; ++j) b[j] = p[j];
#pragma unroll
  for (int j = 0; j < 8; ++j) b[8 + j] = p[16 + j];
  return b;
}

// ---------------------------------------------------------------------------
// generic bf16 WMMA GEMM: C[M,N] = A[M,K] * Bt[N,K]^T (+bias) (ReLU opt)
// REQUIRES: N % 64 == 0, K % 32 == 0 (true for every call in this net).
// block = 128 threads (4 waves); wave tile = 32 rows x 64 cols (8 WMMA / K-step)
// grid = (N/64, ceil(M/128))
// ---------------------------------------------------------------------------
__global__ void gemm_bf16_wmma(const __bf16* __restrict__ A,
                               const __bf16* __restrict__ Bt,
                               const float* __restrict__ bias,
                               float* __restrict__ Cf,
                               __bf16* __restrict__ Cbf,
                               int M, int N, int K, int relu) {
  int wave = threadIdx.x >> 5;
  int row0 = blockIdx.y * 128 + wave * 32;
  int col0 = blockIdx.x * 64;
  if (row0 >= M) return;  // uniform per wave
  f32x8 acc0[4] = {};
  f32x8 acc1[4] = {};
  for (int k0 = 0; k0 < K; k0 += 32) {
    bf16x16 a0 = load_a_frag(A, K, row0, k0, M);
    bf16x16 a1 = load_a_frag(A, K, row0 + 16, k0, M);
#pragma unroll
    for (int t = 0; t < 4; ++t) {
      bf16x16 b = load_b_frag(Bt, K, col0 + t * 16, k0);
      acc0[t] = __builtin_amdgcn_wmma_f32_16x16x32_bf16(
          false, a0, false, b, (short)0, acc0[t], false, false);
      acc1[t] = __builtin_amdgcn_wmma_f32_16x16x32_bf16(
          false, a1, false, b, (short)0, acc1[t], false, false);
    }
  }
  int lane = threadIdx.x & 31;
  int nl = lane & 15, hi = lane >> 4;
#pragma unroll
  for (int t = 0; t < 4; ++t) {
    int c = col0 + t * 16 + nl;
    float bv = bias ? bias[c] : 0.f;
#pragma unroll
    for (int half = 0; half < 2; ++half) {
#pragma unroll
      for (int r = 0; r < 8; ++r) {
        int row = row0 + half * 16 + r + hi * 8;
        if (row < M) {
          float v = (half ? acc1[t][r] : acc0[t][r]) + bv;
          if (relu) v = fmaxf(v, 0.f);
          if (Cf)  Cf[(long)row * N + c] = v;
          if (Cbf) Cbf[(long)row * N + c] = (__bf16)v;
        }
      }
    }
  }
}

// ---------------------------------------------------------------------------
// fused 3-width Conv1d as 15-tap GEMM over padded Xp (B x 526 x 128 bf16)
// Wt: [15][64][128] combined bf16 weights; Ybf: (B*512) x 64 bf16
// grid.y = B*4, block = 128 (4 waves of 32 t-rows x 64 out-ch)
// ---------------------------------------------------------------------------
__global__ void conv_wmma(const __bf16* __restrict__ Xp,
                          const __bf16* __restrict__ Wt,
                          const float* __restrict__ bias,
                          __bf16* __restrict__ Ybf) {
  int wave = threadIdx.x >> 5, lane = threadIdx.x & 31;
  int blk = blockIdx.y;
  int b = blk >> 2;
  int t0 = (blk & 3) * 128 + wave * 32;
  int m = lane & 15, hi = lane >> 4;
  f32x8 acc0[4] = {};
  f32x8 acc1[4] = {};
  for (int d = 0; d < 15; ++d) {
    const __bf16* arow0 = Xp + ((long)(b * 526 + t0 + m + d)) * 128 + hi * 8;
    const __bf16* arow1 = arow0 + 16 * 128;
    const __bf16* wtap = Wt + (long)d * 64 * 128;
#pragma unroll
    for (int kc = 0; kc < 128; kc += 32) {
      bf16x16 a0, a1;
      const __bf16* p0 = arow0 + kc;
      const __bf16* p1 = arow1 + kc;
#pragma unroll
      for (int j = 0; j < 8; ++j) { a0[j] = p0[j]; a1[j] = p1[j]; }
#pragma unroll
      for (int j = 0; j < 8; ++j) { a0[8 + j] = p0[16 + j]; a1[8 + j] = p1[16 + j]; }
#pragma unroll
      for (int t = 0; t < 4; ++t) {
        bf16x16 bb = load_b_frag(wtap, 128, t * 16, kc);
        acc0[t] = __builtin_amdgcn_wmma_f32_16x16x32_bf16(
            false, a0, false, bb, (short)0, acc0[t], false, false);
        acc1[t] = __builtin_amdgcn_wmma_f32_16x16x32_bf16(
            false, a1, false, bb, (short)0, acc1[t], false, false);
      }
    }
  }
  int nl = lane & 15;
#pragma unroll
  for (int t = 0; t < 4; ++t) {
    int c = t * 16 + nl;
    float bv = bias[c];
#pragma unroll
    for (int half = 0; half < 2; ++half) {
#pragma unroll
      for (int r = 0; r < 8; ++r) {
        int row = t0 + half * 16 + r + hi * 8;
        float v = (half ? acc1[t][r] : acc0[t][r]) + bv;
        Ybf[((long)b * 512 + row) * 64 + c] = (__bf16)v;
      }
    }
  }
}

// ---------------------------------------------------------------------------
// small utility kernels
// ---------------------------------------------------------------------------
__global__ void zero_u32_kernel(unsigned* p, long n) {
  long i = (long)blockIdx.x * blockDim.x + threadIdx.x;
  if (i < n) p[i] = 0u;
}
__global__ void cvt_bf16_kernel(const float* a, __bf16* o, long n) {
  long i = (long)blockIdx.x * blockDim.x + threadIdx.x;
  if (i < n) o[i] = (__bf16)a[i];
}
__global__ void transpose_to_bf16(const float* __restrict__ W, __bf16* __restrict__ Bt,
                                  int K, int N) {
  long i = (long)blockIdx.x * blockDim.x + threadIdx.x;
  if (i < (long)K * N) {
    int k = (int)(i / N), n = (int)(i % N);
    Bt[(long)n * K + k] = (__bf16)W[i];
  }
}
__global__ void build_conv_w(const float* c1w, const float* c2w, const float* c3w,
                             __bf16* Wt) {
  int i = blockIdx.x * blockDim.x + threadIdx.x;
  if (i >= 15 * 64 * 128) return;
  int d = i / (64 * 128), rem = i % (64 * 128);
  int o = rem / 128, ci = rem % 128;
  float w = c3w[(o * 128 + ci) * 15 + d];                 // width 15, pad 7
  int d2 = d - 2; if (d2 >= 0 && d2 < 11) w += c2w[(o * 128 + ci) * 11 + d2];
  int d1 = d - 4; if (d1 >= 0 && d1 < 7)  w += c1w[(o * 128 + ci) * 7 + d1];
  Wt[(long)d * 64 * 128 + (long)o * 128 + ci] = (__bf16)(w / 3.f);
}
__global__ void build_conv_b(const float* b1, const float* b2, const float* b3,
                             float* bc) {
  int i = threadIdx.x;
  if (i < 64) bc[i] = (b1[i] + b2[i] + b3[i]) / 3.f;
}
__global__ void embed_kernel(const int* x, const float* e1, const float* e2,
                             float* xr, __bf16* xg, int N) {
  long i = (long)blockIdx.x * blockDim.x + threadIdx.x;
  if (i < (long)N * 128) {
    int n = (int)(i / 128), c = (int)(i % 128);
    int tok = x[n];
    xr[i] = e1[tok * 128 + c];
    xg[i] = (__bf16)e2[tok * 128 + c];
  }
}
__global__ void cumsum_offsets(const int* len, int* off, int B) {
  if (threadIdx.x == 0 && blockIdx.x == 0) {
    int a = 0;
    for (int b = 0; b < B; ++b) { off[b] = a; a += len[b]; }
  }
}

// ---------------------------------------------------------------------------
// GAT attention kernels
// ---------------------------------------------------------------------------
__global__ void attn_dots(const float* __restrict__ h, const float* __restrict__ asrc,
                          const float* __restrict__ adst, float* es, float* ed,
                          int N, int H, int C) {
  int wave = threadIdx.x >> 5, lane = threadIdx.x & 31;
  long wid = (long)blockIdx.x * 4 + wave;
  if (wid >= (long)N * H) return;
  int n = (int)(wid / H), hh = (int)(wid % H);
  const float* hp = h + ((long)n * H + hh) * C;
  const float* as = asrc + hh * C;
  const float* ad = adst + hh * C;
  float s1 = 0.f, s2 = 0.f;
  for (int c = lane; c < C; c += 32) {
    float v = hp[c];
    s1 += v * as[c];
    s2 += v * ad[c];
  }
#pragma unroll
  for (int off = 16; off; off >>= 1) {
    s1 += __shfl_xor(s1, off, 32);
    s2 += __shfl_xor(s2, off, 32);
  }
  if (lane == 0) { es[wid] = s1; ed[wid] = s2; }
}

__global__ void edge_max(const int* ei, long E0, long E, const float* es,
                         const float* ed, unsigned* menc, int H) {
  long i = (long)blockIdx.x * blockDim.x + threadIdx.x;
  if (i >= E * H) return;
  long e = i / H; int hh = (int)(i % H);
  int s, d; edge_sd(ei, E0, e, s, d);
  float v = es[(long)s * H + hh] + ed[(long)d * H + hh];
  v = (v > 0.f) ? v : LEAKY * v;
  atomicMax(&menc[(long)d * H + hh], ord_enc(v));
}

__global__ void edge_expsum(const int* ei, long E0, long E, const float* es,
                            const float* ed, const unsigned* menc, float* z,
                            float* ssum, int H) {
  long i = (long)blockIdx.x * blockDim.x + threadIdx.x;
  if (i >= E * H) return;
  long e = i / H; int hh = (int)(i % H);
  int s, d; edge_sd(ei, E0, e, s, d);
  float v = es[(long)s * H + hh] + ed[(long)d * H + hh];
  v = (v > 0.f) ? v : LEAKY * v;
  float m = ord_dec(menc[(long)d * H + hh]);
  float zz = __expf(v - m);
  z[i] = zz;
  atomicAdd(&ssum[(long)d * H + hh], zz);
}

__global__ void edge_agg(const int* ei, long E0, long E, const float* z,
                         const float* ssum, const float* __restrict__ h,
                         float* agg, int H, int C) {
  int wave = threadIdx.x >> 5, lane = threadIdx.x & 31;
  long wid = (long)blockIdx.x * 4 + wave;
  if (wid >= E * H) return;
  long e = wid / H; int hh = (int)(wid % H);
  int s, d; edge_sd(ei, E0, e, s, d);
  float alpha = z[wid] / (ssum[(long)d * H + hh] + 1e-16f);
  const float* hs = h + ((long)s * H + hh) * C;
  float* ag = agg + ((long)d * H + hh) * C;
  for (int c = lane; c < C; c += 32) atomicAdd(&ag[c], alpha * hs[c]);
}

__global__ void gat_final(const float* agg, const float* bias, float* actf,
                          __bf16* actbf, int N, int H, int C) {
  long i = (long)blockIdx.x * blockDim.x + threadIdx.x;
  if (i >= (long)N * C) return;
  int n = (int)(i / C), c = (int)(i % C);
  float s = 0.f;
  for (int hh = 0; hh < H; ++hh) s += agg[((long)n * H + hh) * C + c];
  float v = fmaxf(s / (float)H + bias[c], 0.f);
  if (actf) actf[i] = v;
  actbf[i] = (__bf16)v;
}

// ---------------------------------------------------------------------------
// graph embedding, gather, sequence embedding
// ---------------------------------------------------------------------------
__global__ void graph_emb_acc(const float* h3, const int* batch, float* eg, int N) {
  long i = (long)blockIdx.x * blockDim.x + threadIdx.x;
  if (i < (long)N * 128) {
    int n = (int)(i / 128), c = (int)(i % 128);
    atomicAdd(&eg[(long)batch[n] * 128 + c], h3[i]);
  }
}
__global__ void graph_emb_div(float* eg, const int* len, int B) {
  int i = blockIdx.x * blockDim.x + threadIdx.x;
  if (i < B * 128) eg[i] /= (float)len[i / 128];
}
__global__ void gather_kernel(const float* h3, const float* xr, const float* embp,
                              const int* off, const int* len, int N,
                              float* out_graph, float* mask1, float* mask2,
                              __bf16* Xp) {
  int p = blockIdx.x;           // b*512 + t
  int b = p >> 9, t = p & 511;
  int c = threadIdx.x;          // 0..127
  int g = off[b] + t;
  if (g < 0) g = 0;
  if (g >= N) g = N - 1;
  float msk = (t < len[b]) ? 1.f : 0.f;
  long oidx = (long)p * 128 + c;
  out_graph[oidx] = h3[(long)g * 128 + c] * msk;
  float orr = (xr[(long)g * 128 + c] * msk + embp[(long)g * 128 + c] * msk) * 0.5f;
  Xp[((long)b * 526 + t + 7) * 128 + c] = (__bf16)orr;
  if (c == 0) { mask1[p] = msk; mask2[p] = msk; }
}
__global__ void emb_seq_kernel(const float* out_cnn, const float* mask,
                               float* es_out, int B) {
  int b = blockIdx.x, c = threadIdx.x;  // 128 threads
  float s = 0.f;
  for (int t = 0; t < 512; ++t)
    s += out_cnn[((long)b * 512 + t) * 128 + c] * mask[b * 512 + t];
  es_out[(long)b * 128 + c] = s / 512.f;
}

// ---------------------------------------------------------------------------
// host launcher
// ---------------------------------------------------------------------------
extern "C" void kernel_launch(void* const* d_in, const int* in_sizes, int n_in,
                              void* d_out, int out_size, void* d_ws, size_t ws_size,
                              hipStream_t stream) {
  const int  N  = in_sizes[0];
  const long E0 = in_sizes[1] / 2;
  const int  B  = in_sizes[3];
  const long E  = E0 + N;

  const int*   x     = (const int*)d_in[0];
  const int*   ei    = (const int*)d_in[1];
  const int*   batch = (const int*)d_in[2];
  const int*   rlen  = (const int*)d_in[3];
  const float* emb   = (const float*)d_in[4];
  const float* emb1  = (const float*)d_in[5];
  const float* emb2  = (const float*)d_in[6];
  const float* w1    = (const float*)d_in[7];
  const float* as1   = (const float*)d_in[8];
  const float* ad1   = (const float*)d_in[9];
  const float* b1    = (const float*)d_in[10];
  const float* w2    = (const float*)d_in[11];
  const float* as2   = (const float*)d_in[12];
  const float* ad2   = (const float*)d_in[13];
  const float* b2    = (const float*)d_in[14];
  const float* w3    = (const float*)d_in[15];
  const float* as3   = (const float*)d_in[16];
  const float* ad3   = (const float*)d_in[17];
  const float* b3    = (const float*)d_in[18];
  const float* lew   = (const float*)d_in[19];
  const float* leb   = (const float*)d_in[20];
  const float* c1w   = (const float*)d_in[21];
  const float* c1b   = (const float*)d_in[22];
  const float* c2w   = (const float*)d_in[23];
  const float* c2b   = (const float*)d_in[24];
  const float* c3w   = (const float*)d_in[25];
  const float* c3b   = (const float*)d_in[26];
  const float* l1w   = (const float*)d_in[27];
  const float* l1b   = (const float*)d_in[28];
  const float* l2w   = (const float*)d_in[29];
  const float* l2b   = (const float*)d_in[30];

  // outputs (concatenated)
  float* out_cnn   = (float*)d_out;
  float* out_graph = out_cnn + (long)B * 512 * 128;
  float* mask1     = out_graph + (long)B * 512 * 128;
  float* mask2     = mask1 + (long)B * 512;
  float* emb_seq   = mask2 + (long)B * 512;
  float* emb_graph = emb_seq + (long)B * 128;

  // workspace bump allocator
  char* ws = (char*)d_ws;
  size_t off = 0;
  auto alloc = [&](size_t bytes) -> void* {
    off = (off + 255) & ~(size_t)255;
    void* p = ws + off;
    off += bytes;
    return p;
  };

  __bf16* w1t   = (__bf16*)alloc((size_t)1024 * 128 * 2);
  __bf16* w2t   = (__bf16*)alloc((size_t)1024 * 256 * 2);
  __bf16* w3t   = (__bf16*)alloc((size_t)128 * 256 * 2);
  __bf16* lewt  = (__bf16*)alloc((size_t)128 * 640 * 2);
  __bf16* l1wt  = (__bf16*)alloc((size_t)512 * 64 * 2);
  __bf16* l2wt  = (__bf16*)alloc((size_t)128 * 512 * 2);
  __bf16* convw = (__bf16*)alloc((size_t)15 * 64 * 128 * 2);
  float*  convb = (float*)alloc(64 * 4);
  float*  xr    = (float*)alloc((size_t)N * 128 * 4);
  __bf16* actbf = (__bf16*)alloc((size_t)N * 256 * 2);   // also holds xg (N x 128)
  __bf16* embbf = (__bf16*)alloc((size_t)N * 640 * 2);
  float*  hpre  = (float*)alloc((size_t)N * 1024 * 4);
  float*  agg   = (float*)alloc((size_t)N * 1024 * 4);
  float*  es    = (float*)alloc((size_t)N * 4 * 4);
  float*  ed    = (float*)alloc((size_t)N * 4 * 4);
  unsigned* menc = (unsigned*)alloc((size_t)N * 4 * 4);
  float*  ssum  = (float*)alloc((size_t)N * 4 * 4);
  float*  zbuf  = (float*)alloc((size_t)E * 4 * 4);
  float*  h3    = (float*)alloc((size_t)N * 128 * 4);
  float*  embp  = (float*)alloc((size_t)N * 128 * 4);
  int*    offs  = (int*)alloc((size_t)B * 4);
  __bf16* Xp    = (__bf16*)alloc((size_t)B * 526 * 128 * 2);
  __bf16* ybf   = (__bf16*)alloc((size_t)B * 512 * 64 * 2);
  __bf16* t1bf  = (__bf16*)alloc((size_t)B * 512 * 512 * 2);

  auto zero = [&](void* p, long words) {
    zero_u32_kernel<<<dim3((unsigned)((words + 255) / 256)), dim3(256), 0, stream>>>(
        (unsigned*)p, words);
  };
  auto gemm = [&](const __bf16* A, const __bf16* Bt, const float* bias, float* Cf,
                  __bf16* Cbf, int M, int Nc, int K, int relu) {
    dim3 g((unsigned)(Nc / 64), (unsigned)((M + 127) / 128));
    gemm_bf16_wmma<<<g, dim3(128), 0, stream>>>(A, Bt, bias, Cf, Cbf, M, Nc, K, relu);
  };

  // --- weight prep ---
  auto tr = [&](const float* W, __bf16* Bt, int K, int Nc) {
    long n = (long)K * Nc;
    transpose_to_bf16<<<dim3((unsigned)((n + 255) / 256)), dim3(256), 0, stream>>>(
        W, Bt, K, Nc);
  };
  tr(w1, w1t, 128, 1024);
  tr(w2, w2t, 256, 1024);
  tr(w3, w3t, 256, 128);
  tr(lew, lewt, 640, 128);
  tr(l1w, l1wt, 64, 512);
  tr(l2w, l2wt, 512, 128);
  build_conv_w<<<dim3((15 * 64 * 128 + 255) / 256), dim3(256), 0, stream>>>(
      c1w, c2w, c3w, convw);
  build_conv_b<<<dim3(1), dim3(64), 0, stream>>>(c1b, c2b, c3b, convb);

  // --- embeddings, offsets ---
  embed_kernel<<<dim3((unsigned)(((long)N * 128 + 255) / 256)), dim3(256), 0, stream>>>(
      x, emb1, emb2, xr, actbf, N);
  cvt_bf16_kernel<<<dim3((unsigned)(((long)N * 640 + 255) / 256)), dim3(256), 0,
                    stream>>>(emb, embbf, (long)N * 640);
  cumsum_offsets<<<dim3(1), dim3(32), 0, stream>>>(rlen, offs, B);

  // --- GAT layers ---
  auto gat_layer = [&](const __bf16* Wt, int K, int H, int C, const float* as_,
                       const float* ad_, const float* bias, float* actf_out) {
    int NC = H * C;
    gemm(actbf, Wt, nullptr, hpre, nullptr, N, NC, K, 0);
    long NH = (long)N * H;
    attn_dots<<<dim3((unsigned)((NH + 3) / 4)), dim3(128), 0, stream>>>(
        hpre, as_, ad_, es, ed, N, H, C);
    zero(menc, NH);
    zero(ssum, NH);
    zero(agg, (long)N * NC);
    long EH = E * H;
    edge_max<<<dim3((unsigned)((EH + 255) / 256)), dim3(256), 0, stream>>>(
        ei, E0, E, es, ed, menc, H);
    edge_expsum<<<dim3((unsigned)((EH + 255) / 256)), dim3(256), 0, stream>>>(
        ei, E0, E, es, ed, menc, zbuf, ssum, H);
    edge_agg<<<dim3((unsigned)((EH + 3) / 4)), dim3(128), 0, stream>>>(
        ei, E0, E, zbuf, ssum, hpre, agg, H, C);
    gat_final<<<dim3((unsigned)(((long)N * C + 255) / 256)), dim3(256), 0, stream>>>(
        agg, bias, actf_out, actbf, N, H, C);
  };
  gat_layer(w1t, 128, 4, 256, as1, ad1, b1, nullptr);
  gat_layer(w2t, 256, 4, 256, as2, ad2, b2, nullptr);
  gat_layer(w3t, 256, 1, 128, as3, ad3, b3, h3);

  // --- graph embedding (segment mean of h3 over batch) ---
  zero(emb_graph, (long)B * 128);
  graph_emb_acc<<<dim3((unsigned)(((long)N * 128 + 255) / 256)), dim3(256), 0,
                  stream>>>(h3, batch, emb_graph, N);
  graph_emb_div<<<dim3((B * 128 + 255) / 256), dim3(256), 0, stream>>>(
      emb_graph, rlen, B);

  // --- emb_p = relu(emb @ lew + leb) ---
  gemm(embbf, lewt, leb, embp, nullptr, N, 128, 640, 1);

  // --- gather + mask + out_r -> padded conv input ---
  zero(Xp, (long)B * 526 * 64);  // bf16 words -> u32 count
  gather_kernel<<<dim3((unsigned)(B * 512)), dim3(128), 0, stream>>>(
      h3, xr, embp, offs, rlen, N, out_graph, mask1, mask2, Xp);

  // --- CNN: fused 3-width conv, then MLP ---
  conv_wmma<<<dim3(1, (unsigned)(B * 4)), dim3(128), 0, stream>>>(Xp, convw, convb,
                                                                  ybf);
  gemm(ybf, l1wt, l1b, nullptr, t1bf, B * 512, 512, 64, 1);
  gemm(t1bf, l2wt, l2b, out_cnn, nullptr, B * 512, 128, 512, 0);

  // --- emb_seq = masked mean over sequence axis ---
  emb_seq_kernel<<<dim3((unsigned)B), dim3(128), 0, stream>>>(out_cnn, mask1,
                                                              emb_seq, B);

  (void)n_in; (void)out_size; (void)ws_size;
}
